// FourierMamba2DTest_28132035788852
// MI455X (gfx1250) — matrix operand order repaired and checked by reading, not compiled, not run on hardware
//
#include <hip/hip_runtime.h>
#include <hip/hip_bf16.h>
#include <stdint.h>

// MI455X / gfx1250, wave32. GEMMs via v_wmma_f32_16x16x32_f16.
// Data movement: double-buffered global_load_async_to_lds_b128 (ASYNCcnt<=4
// partial waits) for most tiles; double-buffered TDM tensor_load_to_lds
// (TENSORcnt<=2) for the big w_in GEMM. Working set << 192MB L2.

typedef __attribute__((ext_vector_type(16))) _Float16 v16h;
typedef __attribute__((ext_vector_type(8)))  _Float16 v8h;
typedef __attribute__((ext_vector_type(8)))  float    v8f;
typedef uint32_t u32x4 __attribute__((ext_vector_type(4)));
typedef uint32_t u32x8 __attribute__((ext_vector_type(8)));

#define L_SEQ 4096   // 64*64 tokens at the finest scale
#define DIM0  48     // model dim d
#define DI    96     // 2*d (mamba inner dim)
#define NST   16     // SSM state size
#define RLOW  3      // dt low-rank = ceil(48/16)

__device__ __forceinline__ float sigmoidf_(float x) { return 1.f / (1.f + __expf(-x)); }

// --- CDNA5 async / TDM helpers ---------------------------------------------

// Per-lane async copy of 16B global -> LDS, tracked by ASYNCcnt (no VGPR data).
__device__ __forceinline__ void async_copy_b128(uint32_t lds_byte, uint64_t gaddr) {
  asm volatile("global_load_async_to_lds_b128 %0, %1, off"
               :: "v"(lds_byte), "v"(gaddr) : "memory");
}
__device__ __forceinline__ void wait_async0() {
  asm volatile("s_wait_asynccnt 0x0" ::: "memory");
}
// ASYNCcnt decrements in issue order: <=4 retires the previous 4-op tile while
// the next tile's 4 ops stay in flight.
__device__ __forceinline__ void wait_async4() {
  asm volatile("s_wait_asynccnt 0x4" ::: "memory");
}

// TDM 2D tile load: rows x tile0_dw dwords from a row-major matrix into LDS.
// Builds D# group0 (128b) + group1 (256b) per CDNA5 ISA
// and issues tensor_load_to_lds with 2 SGPR groups (<=2D tensor).
__device__ __forceinline__ void tdm_load_2d(uint32_t lds_byte, uint64_t ga,
                                            uint32_t tdim0_dw, uint32_t tdim1,
                                            uint32_t tile0_dw, uint32_t tile1,
                                            uint32_t stride0_dw) {
  u32x4 g0;
  g0[0] = 1u;                                    // count=1 valid user descriptor
  g0[1] = lds_byte;                              // lds_addr      [63:32]
  g0[2] = (uint32_t)ga;                          // global_addr   [95:64]
  g0[3] = (uint32_t)((ga >> 32) & 0x01FFFFFFu)   // global_addr   [120:96]
        | (2u << 30);                            // type=2 (image)
  u32x8 g1;
  g1[0] = (2u << 16);                            // data_size=2 (4-byte elements)
  g1[1] = (tdim0_dw & 0xFFFFu) << 16;            // tensor_dim0 [79:48] lo
  g1[2] = ((tdim0_dw >> 16) & 0xFFFFu)           // tensor_dim0 hi
        | ((tdim1 & 0xFFFFu) << 16);             // tensor_dim1 [111:80] lo
  g1[3] = ((tdim1 >> 16) & 0xFFFFu)              // tensor_dim1 hi
        | ((tile0_dw & 0xFFFFu) << 16);          // tile_dim0  [127:112]
  g1[4] = (tile1 & 0xFFFFu);                     // tile_dim1  [143:128], tile_dim2=0
  g1[5] = stride0_dw;                            // tensor_dim0_stride [207:160] lo
  g1[6] = 0u;                                    // stride hi / dim1_stride lo
  g1[7] = 0u;                                    // dim1_stride hi
  asm volatile("tensor_load_to_lds %0, %1" :: "s"(g0), "s"(g1) : "memory");
}

// ---------------------------------------------------------------------------
__global__ void fill_zero_kernel(float* __restrict__ p, int n) {
  int i = blockIdx.x * blockDim.x + threadIdx.x;
  if (i < n) p[i] = 0.f;
}

// 3x3 conv patch-embed: x[3,64,64] -> emb[L=4096, 48] (token-major for GEMMs)
__global__ void embed_conv_kernel(const float* __restrict__ x,
                                  const float* __restrict__ w,
                                  const float* __restrict__ b,
                                  float* __restrict__ out) {
  int idx = blockIdx.x * blockDim.x + threadIdx.x;  // l*48 + c
  if (idx >= L_SEQ * DIM0) return;
  int c = idx % DIM0, l = idx / DIM0;
  int py = l >> 6, px = l & 63;
  float s = b[c];
  for (int ic = 0; ic < 3; ++ic)
    for (int ky = 0; ky < 3; ++ky) {
      int yy = py + ky - 1; if ((unsigned)yy >= 64u) continue;
      for (int kx = 0; kx < 3; ++kx) {
        int xx = px + kx - 1; if ((unsigned)xx >= 64u) continue;
        s += w[((c * 3 + ic) * 3 + ky) * 3 + kx] * x[(ic * 64 + yy) * 64 + xx];
      }
    }
  out[idx] = s;
}

// f32 -> f16 with zero padding to WMMA-friendly shapes. Fully writes dst.
__global__ void pack_f16_kernel(const float* __restrict__ src, int srows, int scols,
                                int lds, int coff,
                                _Float16* __restrict__ dst, int drows, int dcols) {
  int i = blockIdx.x * blockDim.x + threadIdx.x;
  if (i >= drows * dcols) return;
  int r = i / dcols, c = i % dcols;
  float v = (r < srows && c < scols) ? src[(size_t)r * lds + coff + c] : 0.f;
  dst[i] = (_Float16)v;
}

// --- shared GEMM compute core (fragments + WMMA + epilogue) ----------------
// Tile 64(M) x 64(N), 4 wave32 per block, K staged in LDS in steps of 32.
__device__ __forceinline__ void gemm_mma_step(const _Float16* sA, const _Float16* sW,
                                              int wave, int lane, v8f acc[4]) {
  // A fragment, contiguous K order (k = 16*half + s), matching B
  const int arow = wave * 16 + (lane & 15);
  const int kh   = (lane >> 4) * 16;
  union { v16h v; v8h h[2]; } af;
  af.h[0] = *(const v8h*)&sA[arow * 32 + kh];
  af.h[1] = *(const v8h*)&sA[arow * 32 + kh + 8];
  #pragma unroll
  for (int t = 0; t < 4; ++t) {
    const int brow = t * 16 + (lane & 15);
    const int kb   = (lane >> 4) * 16;
    v16h bf = *(const v16h*)&sW[brow * 32 + kb];
    acc[t] = __builtin_amdgcn_wmma_f32_16x16x32_f16(
        false, af.v, false, bf, (short)0, acc[t], false, false);
  }
}

__device__ __forceinline__ void gemm_epilogue(v8f acc[4], int m0, int n0,
                                              int wave, int lane,
                                              const float* bias, int nbias,
                                              float* C, int ldc, int act) {
  #pragma unroll
  for (int t = 0; t < 4; ++t) {
    const int n = n0 + t * 16 + (lane & 15);
    const float bv = (bias != nullptr && n < nbias) ? bias[n] : 0.f;
    #pragma unroll
    for (int r = 0; r < 8; ++r) {
      const int m = m0 + wave * 16 + (lane >> 4) * 8 + r;
      float v = acc[t][r] + bv;
      if (act == 1)      v = v * sigmoidf_(v);            // SiLU
      else if (act == 2) v = __logf(1.f + __expf(v));     // softplus
      else if (act == 3) v = sigmoidf_(v);                // sigmoid
      C[(size_t)m * ldc + n] = v;
    }
  }
}

// Stage one 64x32-half tile pair (A, W) via async LDS DMA: 4 ops per thread.
__device__ __forceinline__ void stage_async_tile(const _Float16* A, const _Float16* W,
                                                 int m0, int n0, int Kp, int k0,
                                                 uint32_t sAb, uint32_t sWb, int tid) {
  #pragma unroll
  for (int q = 0; q < 2; ++q) {
    const int c   = tid + q * 128;          // 0..255 chunk id
    const int row = c >> 2;
    const int h8  = (c & 3) * 8;            // half-index within row
    const uint32_t loff = (uint32_t)(row * 32 + h8) * 2u;
    async_copy_b128(sAb + loff,
                    (uint64_t)(uintptr_t)&A[(size_t)(m0 + row) * Kp + k0 + h8]);
    async_copy_b128(sWb + loff,
                    (uint64_t)(uintptr_t)&W[(size_t)(n0 + row) * Kp + k0 + h8]);
  }
}

// GEMM, double-buffered async-to-LDS staging (ASYNCcnt path).
__launch_bounds__(128)
__global__ void wmma_gemm_kernel(const _Float16* __restrict__ A,
                                 const _Float16* __restrict__ W,
                                 const float* __restrict__ bias, int nbias,
                                 float* __restrict__ C, int ldc,
                                 int Kp, int act) {
  __shared__ __align__(32) _Float16 sA[2][64 * 32];
  __shared__ __align__(32) _Float16 sW[2][64 * 32];
  const int tid  = threadIdx.x;
  const int lane = tid & 31;
  const int wave = tid >> 5;
  const int m0 = blockIdx.y * 64;
  const int n0 = blockIdx.x * 64;
  const uint32_t sAb[2] = { (uint32_t)(uintptr_t)(void*)&sA[0][0],
                            (uint32_t)(uintptr_t)(void*)&sA[1][0] };
  const uint32_t sWb[2] = { (uint32_t)(uintptr_t)(void*)&sW[0][0],
                            (uint32_t)(uintptr_t)(void*)&sW[1][0] };

  v8f acc[4] = {};
  stage_async_tile(A, W, m0, n0, Kp, 0, sAb[0], sWb[0], tid);
  int buf = 0;
  for (int k0 = 0; k0 < Kp; k0 += 32) {
    const bool have_next = (k0 + 32) < Kp;
    if (have_next) {
      // start tile i+1 into the other buffer, then retire tile i only
      stage_async_tile(A, W, m0, n0, Kp, k0 + 32, sAb[buf ^ 1], sWb[buf ^ 1], tid);
      __builtin_prefetch(&A[(size_t)(m0 + (tid >> 1)) * Kp + k0 + 32], 0, 0);
      wait_async4();
    } else {
      wait_async0();
    }
    __syncthreads();
    gemm_mma_step(&sA[buf][0], &sW[buf][0], wave, lane, acc);
    __syncthreads();   // tile i consumed: its buffer may be overwritten next iter
    buf ^= 1;
  }
  gemm_epilogue(acc, m0, n0, wave, lane, bias, nbias, C, ldc, act);
}

// GEMM, double-buffered TDM staging (TENSORcnt path): wave 0 programs two 2D
// tile descriptors per K step; TDM streams the 4KB tiles while WMMAs run.
__launch_bounds__(128)
__global__ void wmma_gemm_tdm_kernel(const _Float16* __restrict__ A,
                                     const _Float16* __restrict__ W,
                                     const float* __restrict__ bias, int nbias,
                                     float* __restrict__ C, int ldc,
                                     int Kp, int act, int M, int Np) {
  __shared__ __align__(32) _Float16 sA[2][64 * 32];
  __shared__ __align__(32) _Float16 sW[2][64 * 32];
  const int tid  = threadIdx.x;
  const int lane = tid & 31;
  const int wave = tid >> 5;
  const int m0 = blockIdx.y * 64;
  const int n0 = blockIdx.x * 64;
  const uint32_t sAb[2] = { (uint32_t)(uintptr_t)(void*)&sA[0][0],
                            (uint32_t)(uintptr_t)(void*)&sA[1][0] };
  const uint32_t sWb[2] = { (uint32_t)(uintptr_t)(void*)&sW[0][0],
                            (uint32_t)(uintptr_t)(void*)&sW[1][0] };
  const uint32_t strideDw = (uint32_t)(Kp >> 1);   // dwords per source row

  v8f acc[4] = {};
  if (wave == 0) {
    tdm_load_2d(sAb[0], (uint64_t)(uintptr_t)&A[(size_t)m0 * Kp],
                strideDw, (uint32_t)(M - m0), 16u, 64u, strideDw);
    tdm_load_2d(sWb[0], (uint64_t)(uintptr_t)&W[(size_t)n0 * Kp],
                strideDw, (uint32_t)(Np - n0), 16u, 64u, strideDw);
  }
  int buf = 0;
  for (int k0 = 0; k0 < Kp; k0 += 32) {
    const bool have_next = (k0 + 32) < Kp;
    if (wave == 0) {  // TDM ops are in-order per wave: cnt<=2 retires tile i
      if (have_next) {
        const uint32_t remDw = (uint32_t)((Kp - k0 - 32) >> 1);
        tdm_load_2d(sAb[buf ^ 1], (uint64_t)(uintptr_t)&A[(size_t)m0 * Kp + k0 + 32],
                    remDw, (uint32_t)(M - m0), 16u, 64u, strideDw);
        tdm_load_2d(sWb[buf ^ 1], (uint64_t)(uintptr_t)&W[(size_t)n0 * Kp + k0 + 32],
                    remDw, (uint32_t)(Np - n0), 16u, 64u, strideDw);
        __builtin_amdgcn_s_wait_tensorcnt((short)2);
      } else {
        __builtin_amdgcn_s_wait_tensorcnt((short)0);
      }
    }
    __syncthreads();
    gemm_mma_step(&sA[buf][0], &sW[buf][0], wave, lane, acc);
    __syncthreads();
    buf ^= 1;
  }
  gemm_epilogue(acc, m0, n0, wave, lane, bias, nbias, C, ldc, act);
}

// causal depthwise conv (width 4) over the token axis + SiLU
__global__ void dwconv_silu_kernel(const float* __restrict__ xz, int ldxz,
                                   const float* __restrict__ cw,
                                   const float* __restrict__ cb,
                                   float* __restrict__ xc) {
  int i = blockIdx.x * blockDim.x + threadIdx.x;
  if (i >= L_SEQ * DI) return;
  int ch = i % DI, l = i / DI;
  float s = cb[ch];
  #pragma unroll
  for (int t = 0; t < 4; ++t) {
    int ll = l - 3 + t;
    if (ll >= 0) s += cw[ch * 4 + t] * xz[(size_t)ll * ldxz + ch];
  }
  xc[i] = s * sigmoidf_(s);
}

// Fused selective scan: h_s = exp(dt*A_s)*h_s + dt*B_s*x ; y = <h,C> + D*x
__global__ void scan_kernel(const float* __restrict__ dt, int lddt,
                            const float* __restrict__ dbl, int lddbl,
                            const float* __restrict__ xc,
                            const float* __restrict__ A_log,
                            const float* __restrict__ Dsk,
                            float* __restrict__ y) {
  int ch = blockIdx.x * blockDim.x + threadIdx.x;
  if (ch >= DI) return;
  float A[NST], h[NST];
  #pragma unroll
  for (int s = 0; s < NST; ++s) { A[s] = -__expf(A_log[ch * NST + s]); h[s] = 0.f; }
  const float dsk = Dsk[ch];
  for (int l = 0; l < L_SEQ; ++l) {
    const float dtv = dt[(size_t)l * lddt + ch];
    const float xv  = xc[(size_t)l * DI + ch];
    const float* bm = &dbl[(size_t)l * lddbl + RLOW];
    const float* cm = bm + NST;
    float acc = 0.f;
    #pragma unroll
    for (int s = 0; s < NST; ++s) {
      h[s] = __expf(dtv * A[s]) * h[s] + dtv * bm[s] * xv;
      acc += h[s] * cm[s];
    }
    y[(size_t)l * DI + ch] = acc + dsk * xv;
  }
}

__global__ void gate_silu_kernel(float* __restrict__ y,
                                 const float* __restrict__ xz, int ldxz, int off) {
  int i = blockIdx.x * blockDim.x + threadIdx.x;
  if (i >= L_SEQ * DI) return;
  int ch = i % DI, l = i / DI;
  float z = xz[(size_t)l * ldxz + off + ch];
  y[i] *= z * sigmoidf_(z);
}

__global__ void accum_kernel(float* __restrict__ acc, const float* __restrict__ d, int n) {
  int i = blockIdx.x * blockDim.x + threadIdx.x;
  if (i < n) acc[i] += d[i];
}

__global__ void finalize_kernel(const float* __restrict__ x,
                                const float* __restrict__ acc, int ldacc,
                                float* __restrict__ out) {
  int i = blockIdx.x * blockDim.x + threadIdx.x;  // c*4096 + l, c<3
  if (i >= 3 * L_SEQ) return;
  int c = i / L_SEQ, l = i % L_SEQ;
  float rain  = acc[(size_t)l * ldacc + c];
  float clean = acc[(size_t)l * ldacc + 3 + c] + x[i] - rain;
  out[i] = clean;
  out[3 * L_SEQ + i] = rain;
}

// ---------------------------------------------------------------------------
extern "C" void kernel_launch(void* const* d_in, const int* in_sizes, int n_in,
                              void* d_out, int out_size, void* d_ws, size_t ws_size,
                              hipStream_t stream) {
  (void)in_sizes; (void)n_in; (void)out_size; (void)ws_size;
  const float* x    = (const float*)d_in[0];
  const float* pe_w = (const float*)d_in[1];
  const float* pe_b = (const float*)d_in[2];

  char* ws = (char*)d_ws;
  size_t off = 0;
  auto carve = [&](size_t bytes) -> char* {
    char* p = ws + off; off += (bytes + 255) & ~(size_t)255; return p;
  };
  float*    emb  = (float*)   carve((size_t)L_SEQ * 48  * 4);
  _Float16* a16  = (_Float16*)carve((size_t)L_SEQ * 96  * 2);
  _Float16* w16  = (_Float16*)carve((size_t)192   * 96  * 2);
  float*    xz   = (float*)   carve((size_t)L_SEQ * 192 * 4);
  float*    xc   = (float*)   carve((size_t)L_SEQ * 96  * 4);
  float*    dbl  = (float*)   carve((size_t)L_SEQ * 64  * 4);
  float*    dtb  = (float*)   carve((size_t)L_SEQ * 128 * 4);
  float*    yb   = (float*)   carve((size_t)L_SEQ * 96  * 4);
  float*    dirb = (float*)   carve((size_t)L_SEQ * 64  * 4);
  float*    accb = (float*)   carve((size_t)L_SEQ * 64  * 4);

  const dim3 B256(256);
  auto g1 = [](int n) { return dim3((n + 255) / 256); };

  auto pack = [&](const float* src, int sr, int sc, int ld, int coff,
                  _Float16* dst, int dr, int dc) {
    pack_f16_kernel<<<g1(dr * dc), B256, 0, stream>>>(src, sr, sc, ld, coff, dst, dr, dc);
  };
  auto gemm = [&](const _Float16* A, const _Float16* W, const float* bias, int nbias,
                  float* C, int ldc, int M, int Np, int Kp, int act) {
    dim3 grid(Np / 64, M / 64);
    wmma_gemm_kernel<<<grid, dim3(128), 0, stream>>>(A, W, bias, nbias, C, ldc, Kp, act);
  };
  auto gemm_tdm = [&](const _Float16* A, const _Float16* W, const float* bias, int nbias,
                      float* C, int ldc, int M, int Np, int Kp, int act) {
    dim3 grid(Np / 64, M / 64);
    wmma_gemm_tdm_kernel<<<grid, dim3(128), 0, stream>>>(A, W, bias, nbias, C, ldc,
                                                         Kp, act, M, Np);
  };

  fill_zero_kernel<<<g1(L_SEQ * 64), B256, 0, stream>>>(accb, L_SEQ * 64);
  embed_conv_kernel<<<g1(L_SEQ * DIM0), B256, 0, stream>>>(x, pe_w, pe_b, emb);

  // 4 scan directions of the first SS2D block (params d_in[3..38], 9 per mamba)
  for (int d4 = 0; d4 < 4; ++d4) {
    const int base = 3 + 9 * d4;
    const float* w_in  = (const float*)d_in[base + 0];  // [192, 48]
    const float* cw    = (const float*)d_in[base + 1];  // [96,1,1,4]
    const float* cb    = (const float*)d_in[base + 2];  // [96]
    const float* w_x   = (const float*)d_in[base + 3];  // [35, 96]
    const float* w_dt  = (const float*)d_in[base + 4];  // [96, 3]
    const float* b_dt  = (const float*)d_in[base + 5];  // [96]
    const float* A_log = (const float*)d_in[base + 6];  // [96, 16]
    const float* Dsk   = (const float*)d_in[base + 7];  // [96]
    const float* w_out = (const float*)d_in[base + 8];  // [48, 96]

    // (1) xz = emb @ w_in^T  [4096 x 192], Kp 48->64 -- TDM-staged GEMM
    pack(emb, L_SEQ, 48, 48, 0, a16, L_SEQ, 64);
    pack(w_in, 192, 48, 48, 0, w16, 192, 64);
    gemm_tdm(a16, w16, nullptr, 0, xz, 192, L_SEQ, 192, 64, 0);

    // (2) xc = silu(causal dwconv(xz[:, :96]))
    dwconv_silu_kernel<<<g1(L_SEQ * DI), B256, 0, stream>>>(xz, 192, cw, cb, xc);

    // (3) dbl = xc @ w_x^T  [4096 x 64] (35 valid), Kp=96 -- async-staged
    pack(xc, L_SEQ, 96, 96, 0, a16, L_SEQ, 96);
    pack(w_x, 35, 96, 96, 0, w16, 64, 96);
    gemm(a16, w16, nullptr, 0, dbl, 64, L_SEQ, 64, 96, 0);

    // (4) dt = softplus(dbl[:, :3] @ w_dt^T + b_dt)  [4096 x 128] (96 valid)
    pack(dbl, L_SEQ, 3, 64, 0, a16, L_SEQ, 32);
    pack(w_dt, 96, 3, 3, 0, w16, 128, 32);
    gemm(a16, w16, b_dt, 96, dtb, 128, L_SEQ, 128, 32, 2);

    // (5) fused selective scan -> yb [4096 x 96]
    scan_kernel<<<dim3(3), dim3(32), 0, stream>>>(dtb, 128, dbl, 64, xc, A_log, Dsk, yb);

    // (6) gate: yb *= silu(xz[:, 96:192])
    gate_silu_kernel<<<g1(L_SEQ * DI), B256, 0, stream>>>(yb, xz, 192, 96);

    // (7) dir = yb @ w_out^T  [4096 x 64] (48 valid), Kp=96 -- async-staged
    pack(yb, L_SEQ, 96, 96, 0, a16, L_SEQ, 96);
    pack(w_out, 48, 96, 96, 0, w16, 64, 96);
    gemm(a16, w16, nullptr, 0, dirb, 64, L_SEQ, 64, 96, 0);

    // (8) accumulate the 4 directions
    accum_kernel<<<g1(L_SEQ * 64), B256, 0, stream>>>(accb, dirb, L_SEQ * 64);
  }

  finalize_kernel<<<g1(3 * L_SEQ), B256, 0, stream>>>(x, accb, 64, (float*)d_out);
}
